// AttentionBlock_44384192037283
// MI455X (gfx1250) — matrix operand order repaired and checked by reading, not compiled, not run on hardware
//
#include <hip/hip_runtime.h>
#include <math.h>

// ---------------------------------------------------------------------------
// MI455X (gfx1250) attention block: gelu -> QKV -> softmax(qk^T/sqrt(d)) -> V
//        -> out-proj + residual.  All GEMMs via v_wmma_f32_16x16x32_bf16.
// B=4, S=2048, D=1024. ~137 GFLOP vs ~84MB minimal traffic -> compute bound ->
// bf16 matrix inputs, fp32 WMMA accumulation.
// GEMM: 128x256 block tile, 8 waves x (64x64), double-buffered LDS fed by
// GLOBAL_LOAD_ASYNC_TO_LDS_B128 (ASYNCcnt) when the toolchain exposes it.
// ---------------------------------------------------------------------------

typedef __bf16 bf16;
typedef __attribute__((ext_vector_type(16))) __bf16 v16bf;
typedef __attribute__((ext_vector_type(8)))  __bf16 v8bf;
typedef __attribute__((ext_vector_type(8)))  float  v8f;
typedef int v4i_ __attribute__((vector_size(16)));   // matches builtin param type

// ----------------- async global->LDS copy (CDNA5 path) ---------------------
#if __has_builtin(__builtin_amdgcn_global_load_async_to_lds_b128)
#define ASYNC_LDS 1
__device__ __forceinline__ void cp16(bf16* lds_dst, const bf16* gsrc) {
    __builtin_amdgcn_global_load_async_to_lds_b128(
        (__attribute__((address_space(1))) v4i_*)const_cast<bf16*>(gsrc),
        (__attribute__((address_space(3))) v4i_*)lds_dst, 0, 0);
}
__device__ __forceinline__ void wait_async() {
#if __has_builtin(__builtin_amdgcn_s_wait_asynccnt)
    __builtin_amdgcn_s_wait_asynccnt(0);
#else
    asm volatile("s_wait_asynccnt 0" ::: "memory");
#endif
}
#else
#define ASYNC_LDS 0
__device__ __forceinline__ void cp16(bf16* lds_dst, const bf16* gsrc) {
    *(v8bf*)lds_dst = *(const v8bf*)gsrc;
}
__device__ __forceinline__ void wait_async() {}
#endif

__device__ __forceinline__ bf16 f2bf(float f) {
    unsigned u = __builtin_bit_cast(unsigned, f);
    unsigned r = (u + 0x7FFFu + ((u >> 16) & 1u)) >> 16;   // round-nearest-even
    unsigned short s = (unsigned short)r;
    return __builtin_bit_cast(bf16, s);
}

// ------------------------- problem-size constants --------------------------
static constexpr int B_  = 4;
static constexpr int S_  = 2048;
static constexpr int D_  = 1024;
static constexpr int BS_ = B_ * S_;                    // 8192 rows

// ------------------------- workspace layout (bytes) ------------------------
static constexpr size_t SZ_ROWS_BF = (size_t)BS_ * D_ * 2;       // 16 MiB
static constexpr size_t SZ_W_BF    = (size_t)D_ * D_ * 2;        //  2 MiB
static constexpr size_t SZ_SC_F32  = (size_t)B_ * S_ * S_ * 4;   // 64 MiB
static constexpr size_t SZ_SC_BF   = (size_t)B_ * S_ * S_ * 2;   // 32 MiB

static constexpr size_t OFF_XH  = 0;
static constexpr size_t OFF_WQT = OFF_XH  + SZ_ROWS_BF;
static constexpr size_t OFF_WKT = OFF_WQT + SZ_W_BF;
static constexpr size_t OFF_WVT = OFF_WKT + SZ_W_BF;
static constexpr size_t OFF_WOT = OFF_WVT + SZ_W_BF;
static constexpr size_t OFF_QH  = OFF_WOT + SZ_W_BF;
static constexpr size_t OFF_KH  = OFF_QH  + SZ_ROWS_BF;
static constexpr size_t OFF_VH  = OFF_KH  + SZ_ROWS_BF;
static constexpr size_t OFF_VT  = OFF_VH  + SZ_ROWS_BF;
static constexpr size_t OFF_SC  = OFF_VT  + SZ_ROWS_BF;
static constexpr size_t OFF_AT  = OFF_SC  + SZ_SC_F32;
static constexpr size_t OFF_MX  = OFF_AT  + SZ_SC_BF;

// ---------------------------------------------------------------------------
// Generic WMMA GEMM:  C[m][n] = alpha * sum_k A[m][k] * Bt[n][k]  (+bias)(+res)
// A, Bt bf16 row-major. 128(M)x256(N) block tile, 256 threads (8 waves),
// K-step 32, double-buffered LDS. Wave grid 2x4 -> each wave owns 64x64.
// ---------------------------------------------------------------------------

// stage one 128x32 A-tile and 256x32 B-tile into an LDS buffer (async)
__device__ __forceinline__ void stage_tile(bf16 (*As)[48], bf16 (*Bs)[48],
                                           const bf16* __restrict__ A,
                                           const bf16* __restrict__ Bt,
                                           int m0, int n0, int lda, int ldb,
                                           int k0, int t)
{
#pragma unroll
    for (int i = 0; i < 2; ++i) {              // 512 chunks of 8 bf16 (A)
        const int c = t + i * 256;
        const int r = c >> 2, col = (c & 3) << 3;
        cp16(&As[r][col], A + (size_t)(m0 + r) * lda + k0 + col);
    }
#pragma unroll
    for (int i = 0; i < 4; ++i) {              // 1024 chunks of 8 bf16 (B)
        const int c = t + i * 256;
        const int r = c >> 2, col = (c & 3) << 3;
        cp16(&Bs[r][col], Bt + (size_t)(n0 + r) * ldb + k0 + col);
    }
}

__global__ __launch_bounds__(256)
void gemm_wmma_bf16(const bf16* __restrict__ A,  long long sA, int lda,
                    const bf16* __restrict__ Bt, long long sB, int ldb,
                    int K, float alpha,
                    const float* __restrict__ bias,
                    const float* __restrict__ residual, long long sR, int ldr,
                    float* __restrict__ outF, bf16* __restrict__ outH,
                    long long sO, int ldo)
{
    const int t    = threadIdx.x;
    const int bz   = blockIdx.z;
    const int m0   = blockIdx.y * 128;
    const int n0   = blockIdx.x * 256;
    A  += (long long)bz * sA;
    Bt += (long long)bz * sB;

    // row stride 48 elems = 96B: 16B/32B aligned fragment loads, spreads banks
    __shared__ bf16 As[2][128][48];            // 24 KiB
    __shared__ bf16 Bs[2][256][48];            // 48 KiB

    const int wave = t >> 5, lane = t & 31;
    const int wm   = (wave >> 2) * 64;         // 0 | 64
    const int wn   = (wave & 3) * 64;          // 0,64,128,192
    const int lrow = lane & 15;
    const int klA  = (lane >> 4) * 16;         // A K 0..15 | 16..31 per lane half
    const int klB  = (lane >> 4) * 16;         // B K 0..15 | 16..31 per lane half

    v8f acc[4][4];
#pragma unroll
    for (int i = 0; i < 4; ++i)
#pragma unroll
        for (int j = 0; j < 4; ++j)
#pragma unroll
            for (int e = 0; e < 8; ++e) acc[i][j][e] = 0.0f;

    stage_tile(As[0], Bs[0], A, Bt, m0, n0, lda, ldb, 0, t);

    int cur = 0;
    for (int k0 = 0; k0 < K; k0 += 32, cur ^= 1) {
        wait_async();              // this wave's copies into As/Bs[cur] done
        __syncthreads();           // every wave's copies done, prior reads done

        if (k0 + 32 < K)           // overlap next stage with this tile's WMMAs
            stage_tile(As[cur ^ 1], Bs[cur ^ 1], A, Bt, m0, n0, lda, ldb,
                       k0 + 32, t);

        v16bf af[4], bfr[4];
#pragma unroll
        for (int i = 0; i < 4; ++i) {
            v8bf lo = *(const v8bf*)&As[cur][wm + i * 16 + lrow][klA];
            v8bf hi = *(const v8bf*)&As[cur][wm + i * 16 + lrow][8 + klA];
            af[i] = __builtin_shufflevector(lo, hi, 0, 1, 2, 3, 4, 5, 6, 7,
                                                    8, 9, 10, 11, 12, 13, 14, 15);
        }
#pragma unroll
        for (int j = 0; j < 4; ++j)
            bfr[j] = *(const v16bf*)&Bs[cur][wn + j * 16 + lrow][klB];

#pragma unroll
        for (int i = 0; i < 4; ++i)
#pragma unroll
            for (int j = 0; j < 4; ++j)
                acc[i][j] = __builtin_amdgcn_wmma_f32_16x16x32_bf16(
                    false, af[i], false, bfr[j], (short)0, acc[i][j], false, false);
    }

    // Epilogue. C layout: VGPR r -> M = r + 8*(lane>=16); N = lane&15.
    const int mhi = (lane >> 4) * 8;
#pragma unroll
    for (int i = 0; i < 4; ++i) {
#pragma unroll
        for (int j = 0; j < 4; ++j) {
            const int n  = n0 + wn + j * 16 + lrow;
            const float bvl = bias ? bias[n] : 0.0f;
#pragma unroll
            for (int r = 0; r < 8; ++r) {
                const int m = m0 + wm + i * 16 + mhi + r;
                float v = acc[i][j][r] * alpha + bvl;
                if (residual)
                    v += residual[(long long)bz * sR + (size_t)m * ldr + n];
                const size_t o = (size_t)((long long)bz * sO) + (size_t)m * ldo + n;
                if (outF) outF[o] = v;
                else      outH[o] = f2bf(v);
            }
        }
    }
}

// -------------------- exact GELU (erf) + bf16 convert ----------------------
__global__ __launch_bounds__(256)
void gelu_to_bf16(const float* __restrict__ in, bf16* __restrict__ out, int n)
{
    int i0 = (blockIdx.x * 256 + threadIdx.x) * 4;
    if (i0 + 3 >= n) return;
#pragma unroll
    for (int i = 0; i < 4; ++i) {
        float x = in[i0 + i];
        float g = 0.5f * x * (1.0f + erff(x * 0.70710678118654752f));
        out[i0 + i] = f2bf(g);
    }
}

// ------------------ weight: fp32 [k][n] -> bf16 Wt[n][k] -------------------
__global__ __launch_bounds__(256)
void transpose_f32_to_bf16(const float* __restrict__ in, bf16* __restrict__ out,
                           int rows, int cols)
{
    __shared__ float tile[32][33];
    const int c0 = blockIdx.x * 32, r0 = blockIdx.y * 32;
    const int tx = threadIdx.x & 31, ty = threadIdx.x >> 5;
#pragma unroll
    for (int i = 0; i < 32; i += 8)
        tile[ty + i][tx] = in[(size_t)(r0 + ty + i) * cols + c0 + tx];
    __syncthreads();
#pragma unroll
    for (int i = 0; i < 32; i += 8)
        out[(size_t)(c0 + ty + i) * rows + r0 + tx] = f2bf(tile[tx][ty + i]);
}

// -------------------- bf16 [s][d] -> bf16 [d][s], batched ------------------
__global__ __launch_bounds__(256)
void transpose_bf16_batched(const bf16* __restrict__ in, bf16* __restrict__ out,
                            int rows, int cols)
{
    const size_t boff = (size_t)blockIdx.z * rows * cols;
    __shared__ bf16 tile[32][33];
    const int c0 = blockIdx.x * 32, r0 = blockIdx.y * 32;
    const int tx = threadIdx.x & 31, ty = threadIdx.x >> 5;
#pragma unroll
    for (int i = 0; i < 32; i += 8)
        tile[ty + i][tx] = in[boff + (size_t)(r0 + ty + i) * cols + c0 + tx];
    __syncthreads();
#pragma unroll
    for (int i = 0; i < 32; i += 8)
        out[boff + (size_t)(c0 + ty + i) * rows + r0 + tx] = tile[tx][ty + i];
}

// ------------- row softmax: fp32 [rows][2048] -> bf16 probs ----------------
__global__ __launch_bounds__(256)
void softmax_rows(const float* __restrict__ s, bf16* __restrict__ p, int ncols)
{
    const size_t row = blockIdx.x;
    s += row * ncols;
    p += row * ncols;
    __shared__ float red[8];
    const int t = threadIdx.x;

    float vals[8];
    float vmax = -INFINITY;
#pragma unroll
    for (int i = 0; i < 8; ++i) {
        vals[i] = s[t + i * 256];
        vmax = fmaxf(vmax, vals[i]);
    }
#pragma unroll
    for (int off = 16; off > 0; off >>= 1)
        vmax = fmaxf(vmax, __shfl_xor(vmax, off, 32));
    if ((t & 31) == 0) red[t >> 5] = vmax;
    __syncthreads();
    float m = red[0];
#pragma unroll
    for (int w = 1; w < 8; ++w) m = fmaxf(m, red[w]);

    float sum = 0.0f;
#pragma unroll
    for (int i = 0; i < 8; ++i) {
        vals[i] = expf(vals[i] - m);
        sum += vals[i];
    }
#pragma unroll
    for (int off = 16; off > 0; off >>= 1)
        sum += __shfl_xor(sum, off, 32);
    __syncthreads();
    if ((t & 31) == 0) red[t >> 5] = sum;
    __syncthreads();
    float tot = 0.0f;
#pragma unroll
    for (int w = 0; w < 8; ++w) tot += red[w];
    const float inv = 1.0f / tot;
#pragma unroll
    for (int i = 0; i < 8; ++i)
        p[t + i * 256] = f2bf(vals[i] * inv);
}

// ---------------------------------------------------------------------------
extern "C" void kernel_launch(void* const* d_in, const int* in_sizes, int n_in,
                              void* d_out, int out_size, void* d_ws, size_t ws_size,
                              hipStream_t stream)
{
    (void)in_sizes; (void)n_in; (void)out_size; (void)ws_size;

    const float* tokens = (const float*)d_in[0];
    const float* Wq = (const float*)d_in[1];
    const float* bq = (const float*)d_in[2];
    const float* Wk = (const float*)d_in[3];
    const float* bk = (const float*)d_in[4];
    const float* Wv = (const float*)d_in[5];
    const float* bv = (const float*)d_in[6];
    const float* Wo = (const float*)d_in[7];
    const float* bo = (const float*)d_in[8];

    char* ws = (char*)d_ws;
    bf16*  xh     = (bf16*)(ws + OFF_XH);
    bf16*  WqT    = (bf16*)(ws + OFF_WQT);
    bf16*  WkT    = (bf16*)(ws + OFF_WKT);
    bf16*  WvT    = (bf16*)(ws + OFF_WVT);
    bf16*  WoT    = (bf16*)(ws + OFF_WOT);
    bf16*  qh     = (bf16*)(ws + OFF_QH);
    bf16*  kh     = (bf16*)(ws + OFF_KH);
    bf16*  vh     = (bf16*)(ws + OFF_VH);
    bf16*  vT     = (bf16*)(ws + OFF_VT);
    float* scores = (float*)(ws + OFF_SC);
    bf16*  attn   = (bf16*)(ws + OFF_AT);
    bf16*  mixed  = (bf16*)(ws + OFF_MX);
    float* out    = (float*)d_out;

    // 1) weights -> bf16, transposed to [n][k]
    dim3 wgrid(D_ / 32, D_ / 32, 1);
    transpose_f32_to_bf16<<<wgrid, 256, 0, stream>>>(Wq, WqT, D_, D_);
    transpose_f32_to_bf16<<<wgrid, 256, 0, stream>>>(Wk, WkT, D_, D_);
    transpose_f32_to_bf16<<<wgrid, 256, 0, stream>>>(Wv, WvT, D_, D_);
    transpose_f32_to_bf16<<<wgrid, 256, 0, stream>>>(Wo, WoT, D_, D_);

    // 2) x = gelu(tokens) in bf16
    const int nElem = BS_ * D_;
    gelu_to_bf16<<<nElem / (256 * 4), 256, 0, stream>>>(tokens, xh, nElem);

    // 3) Q/K/V projections: [8192,1024] x [1024,1024] + bias -> bf16
    const long long sW = (long long)S_ * D_;
    dim3 qkvGrid(D_ / 256, BS_ / 128, 1);
    gemm_wmma_bf16<<<qkvGrid, 256, 0, stream>>>(xh, 0, D_, WqT, 0, D_, D_, 1.0f,
                                                bq, nullptr, 0, 0, nullptr, qh, 0, D_);
    gemm_wmma_bf16<<<qkvGrid, 256, 0, stream>>>(xh, 0, D_, WkT, 0, D_, D_, 1.0f,
                                                bk, nullptr, 0, 0, nullptr, kh, 0, D_);
    gemm_wmma_bf16<<<qkvGrid, 256, 0, stream>>>(xh, 0, D_, WvT, 0, D_, D_, 1.0f,
                                                bv, nullptr, 0, 0, nullptr, vh, 0, D_);

    // 4) scores[b] = (1/sqrt(D)) * q[b] @ k[b]^T  (k row-major == Bt layout)
    dim3 scGrid(S_ / 256, S_ / 128, B_);
    gemm_wmma_bf16<<<scGrid, 256, 0, stream>>>(qh, sW, D_, kh, sW, D_, D_, 0.03125f,
                                               nullptr, nullptr, 0, 0,
                                               scores, nullptr, (long long)S_ * S_, S_);

    // 5) softmax over keys -> bf16 probs
    softmax_rows<<<B_ * S_, 256, 0, stream>>>(scores, attn, S_);

    // 6) v -> vT[d][s] so attn @ v fits the Bt convention
    dim3 tvGrid(D_ / 32, S_ / 32, B_);
    transpose_bf16_batched<<<tvGrid, 256, 0, stream>>>(vh, vT, S_, D_);

    // 7) mixed[b] = attn[b] @ v[b]
    dim3 avGrid(D_ / 256, S_ / 128, B_);
    gemm_wmma_bf16<<<avGrid, 256, 0, stream>>>(attn, (long long)S_ * S_, S_,
                                               vT, sW, S_, S_, 1.0f,
                                               nullptr, nullptr, 0, 0,
                                               nullptr, mixed, sW, D_);

    // 8) out = mixed @ Wo + bo + tokens  (fp32 output)
    gemm_wmma_bf16<<<qkvGrid, 256, 0, stream>>>(mixed, 0, D_, WoT, 0, D_, D_, 1.0f,
                                                bo, tokens, 0, D_,
                                                out, nullptr, 0, D_);
}
